// EpochMixer_71975061946741
// MI455X (gfx1250) — compile-verified, matrix operand order checked by reading, not run on hardware
//
#include <hip/hip_runtime.h>
#include <stdint.h>

// ---------------- CDNA5 WMMA types ----------------
typedef __attribute__((ext_vector_type(16))) __bf16 v16bf;
typedef __attribute__((ext_vector_type(8)))  float  v8f;

struct Bits256 { uint4 lo, hi; };

static __device__ __forceinline__ uint16_t f2bf(float f) {
  uint32_t u = __builtin_bit_cast(uint32_t, f);
  u += 0x7FFFu + ((u >> 16) & 1u);          // round-to-nearest-even
  return (uint16_t)(u >> 16);
}
static __device__ __forceinline__ float bf2f(uint16_t h) {
  uint32_t u = (uint32_t)h << 16;
  return __builtin_bit_cast(float, u);
}

// Packed f32 -> bf16x2 conversion (CDNA5 v_cvt_pk_bf16_f32, HW RNE).
// Result: [15:0] = bf16(lo), [31:16] = bf16(hi).
static __device__ __forceinline__ uint32_t cvt_pk_bf16(float lo, float hi) {
  uint32_t d;
  asm("v_cvt_pk_bf16_f32 %0, %1, %2" : "=v"(d) : "v"(lo), "v"(hi));
  return d;
}

// Byte offset of an LDS pointer within the workgroup's LDS allocation.
static __device__ __forceinline__ uint32_t lds_offset(const void* p) {
  return (uint32_t)(uintptr_t)(__attribute__((address_space(3))) const void*)p;
}

// Load a 16-element bf16 fragment (A or B operand): two contiguous 16B chunks
// at p and p+16 elements, matching the ISA 16-bit operand layout
// (lane half selects K base 0/8; second run at K+16).
static __device__ __forceinline__ v16bf load_frag(const uint16_t* p) {
  Bits256 t;
  t.lo = *(const uint4*)(p);
  t.hi = *(const uint4*)(p + 16);
  return __builtin_bit_cast(v16bf, t);
}

// acc += A(16 x Ka, LDS row-major) * B^T with W[N x Kb] row-major in LDS.
static __device__ __forceinline__ v8f gemm_acc(const uint16_t* __restrict__ A, int Ka,
                                               const uint16_t* __restrict__ W, int Kb,
                                               int n0, v8f acc, int lrow, int koff) {
  const uint16_t* arow = A + lrow * Ka + koff;
  const uint16_t* wrow = W + (size_t)(n0 + lrow) * Kb + koff;
  #pragma unroll
  for (int ks = 0; ks < 4; ++ks) {
    v16bf a = load_frag(arow + ks * 32);
    v16bf b = load_frag(wrow + ks * 32);
    acc = __builtin_amdgcn_wmma_f32_16x16x32_bf16(false, a, false, b,
                                                  (short)0, acc, false, false);
  }
  return acc;
}

static __device__ __forceinline__ void wait_ds() {
  asm volatile("s_wait_dscnt 0" ::: "memory");
}
static __device__ __forceinline__ void wait_async() {
  asm volatile("s_wait_asynccnt 0" ::: "memory");
}

// Issue async global->LDS staging of a 2D bf16 tile (ASYNCcnt tracked, no
// wait) -- all 256 threads cooperate. rowBytes/srcStrideBytes multiples of 16.
static __device__ __forceinline__ void stage_issue(uint16_t* dst, const uint16_t* src,
                                                   int rows, int rowBytes,
                                                   int srcStrideBytes, int tid) {
  const uint32_t dbase = lds_offset(dst);
  const int chunksPerRow = rowBytes >> 4;
  const int total = rows * chunksPerRow;
  for (int c = tid; c < total; c += 256) {
    const int r = c / chunksPerRow;
    const int k = c - r * chunksPerRow;
    const uint32_t doff = dbase + (uint32_t)(r * rowBytes + k * 16);
    const uint32_t soff = (uint32_t)(r * srcStrideBytes + k * 16);
    asm volatile("global_load_async_to_lds_b128 %0, %1, %2"
                 :: "v"(doff), "v"(soff), "s"(src) : "memory");
  }
}

// LayerNorm over 16 rows x 128 cols: 2 lanes per row, ds_swizzle SWAPX16
// combine; rewrites f32 activations and the bf16 A-staging mirror.
static __device__ __forceinline__ void layer_norm_rows(float* __restrict__ xf,
                                                       uint16_t* __restrict__ xb,
                                                       const float* __restrict__ g,
                                                       const float* __restrict__ b,
                                                       int lane) {
  const int row = lane & 15;
  const int c0  = (lane >> 4) * 64;
  float sum = 0.f, sq = 0.f;
  #pragma unroll
  for (int i = 0; i < 16; ++i) {
    float4 v = *(const float4*)(xf + row * 128 + c0 + i * 4);
    sum += v.x + v.y + v.z + v.w;
    sq  += v.x * v.x + v.y * v.y + v.z * v.z + v.w * v.w;
  }
  sum += __builtin_bit_cast(float,
           __builtin_amdgcn_ds_swizzle(__builtin_bit_cast(int, sum), 0x401F));
  sq  += __builtin_bit_cast(float,
           __builtin_amdgcn_ds_swizzle(__builtin_bit_cast(int, sq), 0x401F));
  const float mu   = sum * (1.f / 128.f);
  const float var  = sq * (1.f / 128.f) - mu * mu;
  const float rstd = __frsqrt_rn(var + 1e-5f);
  #pragma unroll
  for (int i = 0; i < 16; ++i) {
    float4 v = *(const float4*)(xf + row * 128 + c0 + i * 4);
    float y0 = (v.x - mu) * rstd * g[c0 + i * 4 + 0] + b[c0 + i * 4 + 0];
    float y1 = (v.y - mu) * rstd * g[c0 + i * 4 + 1] + b[c0 + i * 4 + 1];
    float y2 = (v.z - mu) * rstd * g[c0 + i * 4 + 2] + b[c0 + i * 4 + 2];
    float y3 = (v.w - mu) * rstd * g[c0 + i * 4 + 3] + b[c0 + i * 4 + 3];
    *(float4*)(xf + row * 128 + c0 + i * 4) = make_float4(y0, y1, y2, y3);
    uint2 p;
    p.x = cvt_pk_bf16(y0, y1);
    p.y = cvt_pk_bf16(y2, y3);
    *(uint2*)(xb + row * 128 + c0 + i * 4) = p;
  }
}

#define WAVES 8

// LDS layout (bytes):
//   [0,       98304)  wstage : shared staged weights (Wqkv whole / Wo / W1q+W2q)
//   [98304,  163840)  xf     : 8 waves x 16x128 f32
//   [163840, 196608)  xb     : 8 waves x 16x128 bf16 (A staging)
//   [196608, 294912)  qkv    : 8 waves x 16x384 bf16
//   [294912, 303104)  hb     : 8 waves x 16x32  bf16
#define SMEM_BYTES 303104

__global__ __launch_bounds__(256) void epoch_mixer_kernel(
    const float* __restrict__ z0, const float* __restrict__ z1,
    const float* __restrict__ z2, const float* __restrict__ cls,
    const uint16_t* __restrict__ WqkvB, const float* __restrict__ bqkv,
    const uint16_t* __restrict__ WoB,   const float* __restrict__ bo,
    const uint16_t* __restrict__ W1B,   const float* __restrict__ b1,
    const uint16_t* __restrict__ W2B,   const float* __restrict__ b2,
    const float* __restrict__ ln1_g, const float* __restrict__ ln1_b,
    const float* __restrict__ ln2_g, const float* __restrict__ ln2_b,
    float* __restrict__ out) {
  extern __shared__ char smem[];
  const int tid   = threadIdx.x;
  const int wave  = tid >> 5;
  const int lane  = tid & 31;
  const int lrow  = lane & 15;
  const int half  = lane >> 4;
  const int koff  = half * 8;   // bf16 element offset of first K run
  const int rbase = half * 8;   // C/D row base for this lane half

  uint16_t* wstage = (uint16_t*)smem;                                              // 96 KB
  float*    xf  = (float*)(smem + 98304) + (size_t)wave * 16 * 128;
  uint16_t* xb  = (uint16_t*)(smem + 163840) + (size_t)wave * 16 * 128;
  uint16_t* qkv = (uint16_t*)(smem + 196608) + (size_t)wave * 16 * 384;
  uint16_t* hb  = (uint16_t*)(smem + 294912) + (size_t)wave * 16 * 32;

  const int seq0 = (blockIdx.x * WAVES + wave) * 4;   // 4 sequences per wave

  // Issue layer-0 Wqkv staging immediately; it overlaps the x-tile build.
  stage_issue(wstage, WqkvB, 1, 98304, 98304, tid);

  // ---------------- build initial 16x128 activation tile ----------------
  {
    const int c = lane * 4;
    #pragma unroll
    for (int r = 0; r < 16; ++r) {
      const int s = r >> 2, tok = r & 3;
      const float* src;
      if (tok == 0) {
        src = cls + c;
      } else {
        const float* z = (tok == 1) ? z0 : (tok == 2 ? z1 : z2);
        src = z + (size_t)(seq0 + s) * 128 + c;
      }
      float4 v = *(const float4*)src;
      *(float4*)(xf + r * 128 + c) = v;
      uint2 p;
      p.x = cvt_pk_bf16(v.x, v.y);
      p.y = cvt_pk_bf16(v.z, v.w);
      *(uint2*)(xb + r * 128 + c) = p;
    }
  }
  wait_ds();
  wait_async();
  __syncthreads();

  for (int l = 0; l < 2; ++l) {
    // ---------------- QKV projection: 16x384 = (16x128)*(128x384) ----------------
    const float* bq = bqkv + l * 384;
    for (int nt = 0; nt < 24; ++nt) {
      const float bv = bq[nt * 16 + lrow];
      v8f acc;
      #pragma unroll
      for (int j = 0; j < 8; ++j) acc[j] = bv;
      acc = gemm_acc(xb, 128, wstage, 128, nt * 16, acc, lrow, koff);
      const int col = nt * 16 + lrow;
      #pragma unroll
      for (int j = 0; j < 8; j += 2) {
        uint32_t p = cvt_pk_bf16(acc[j], acc[j + 1]);
        qkv[(rbase + j) * 384 + col]     = (uint16_t)p;
        qkv[(rbase + j + 1) * 384 + col] = (uint16_t)(p >> 16);
      }
    }
    wait_ds();

    // All waves done reading Wqkv; stage Wo async, shadowed by attention.
    __syncthreads();
    stage_issue(wstage, WoB + (size_t)l * 128 * 128, 1, 32768, 32768, tid);

    // ---------------- attention: one (seq, head) pair per lane ----------------
    {
      const int s = lane >> 3, h = lane & 7;
      const uint16_t* base = qkv + (size_t)(s * 4) * 384 + h * 16;
      #pragma unroll
      for (int qr = 0; qr < 4; ++qr) {
        float qv[16];
        #pragma unroll
        for (int d = 0; d < 16; ++d) qv[d] = bf2f(base[qr * 384 + d]);
        float sc[4]; float m = -1e30f;
        #pragma unroll
        for (int kr = 0; kr < 4; ++kr) {
          float a = 0.f;
          #pragma unroll
          for (int d = 0; d < 16; ++d)
            a += qv[d] * bf2f(base[kr * 384 + 128 + d]);
          sc[kr] = a * 0.25f;           // 1/sqrt(dh), dh=16
          m = fmaxf(m, sc[kr]);
        }
        float den = 0.f;
        #pragma unroll
        for (int kr = 0; kr < 4; ++kr) { sc[kr] = __expf(sc[kr] - m); den += sc[kr]; }
        const float inv = 1.f / den;
        float ov[16];
        #pragma unroll
        for (int d = 0; d < 16; ++d) {
          float o = 0.f;
          #pragma unroll
          for (int kr = 0; kr < 4; ++kr)
            o += sc[kr] * bf2f(base[kr * 384 + 256 + d]);
          ov[d] = o * inv;
        }
        uint16_t* orow = xb + (s * 4 + qr) * 128 + h * 16;
        #pragma unroll
        for (int d = 0; d < 16; d += 2)
          *(uint32_t*)(orow + d) = cvt_pk_bf16(ov[d], ov[d + 1]);
      }
    }
    wait_ds();
    wait_async();      // Wo staged
    __syncthreads();

    // ---------------- output projection + residual ----------------
    const float* bo_ = bo + l * 128;
    for (int nt = 0; nt < 8; ++nt) {
      const float bv = bo_[nt * 16 + lrow];
      v8f acc;
      #pragma unroll
      for (int j = 0; j < 8; ++j) acc[j] = bv;
      acc = gemm_acc(xb, 128, wstage, 128, nt * 16, acc, lrow, koff);
      #pragma unroll
      for (int j = 0; j < 8; ++j)
        xf[(rbase + j) * 128 + nt * 16 + lrow] += acc[j];   // residual add
    }
    wait_ds();
    layer_norm_rows(xf, xb, ln1_g + l * 128, ln1_b + l * 128, lane);
    wait_ds();

    // ---------------- FFN: h = GELU(x W1^T + b1); ff = h W2^T + b2 ----------------
    const float* b1_ = b1 + l * 512;
    const float* b2_ = b2 + l * 128;
    uint16_t* w1s = wstage;             // [128 hidden rows][128 k] bf16 = 32 KB
    uint16_t* w2s = wstage + 16384;     // [128 out rows][128 k-slice] bf16 = 32 KB
    v8f ff[8];
    #pragma unroll
    for (int nt = 0; nt < 8; ++nt) {
      const float bv = b2_[nt * 16 + lrow];
      #pragma unroll
      for (int j = 0; j < 8; ++j) ff[nt][j] = bv;
    }
    for (int g = 0; g < 4; ++g) {       // 512 hidden dims in 4 quarters of 128
      __syncthreads();                  // previous wstage consumers done
      stage_issue(w1s, W1B + (size_t)l * 512 * 128 + (size_t)g * 128 * 128,
                  1, 32768, 32768, tid);                      // contiguous rows
      stage_issue(w2s, W2B + (size_t)l * 128 * 512 + (size_t)g * 128,
                  128, 256, 1024, tid);                       // strided k-slice
      wait_async();
      __syncthreads();
      #pragma unroll
      for (int kcl = 0; kcl < 4; ++kcl) {   // 128 hidden dims, 32 at a time
        const int kc = g * 4 + kcl;
        #pragma unroll
        for (int t = 0; t < 2; ++t) {
          const float bv = b1_[kc * 32 + t * 16 + lrow];
          v8f h;
          #pragma unroll
          for (int j = 0; j < 8; ++j) h[j] = bv;
          h = gemm_acc(xb, 128, w1s, 128, kcl * 32 + t * 16, h, lrow, koff);
          const int col = t * 16 + lrow;
          #pragma unroll
          for (int j = 0; j < 8; j += 2) {
            float v0 = h[j], v1 = h[j + 1];
            v0 = 0.5f * v0 * (1.f + erff(v0 * 0.70710678118f));   // exact GELU
            v1 = 0.5f * v1 * (1.f + erff(v1 * 0.70710678118f));
            uint32_t p = cvt_pk_bf16(v0, v1);
            hb[(rbase + j) * 32 + col]     = (uint16_t)p;
            hb[(rbase + j + 1) * 32 + col] = (uint16_t)(p >> 16);
          }
        }
        wait_ds();
        const v16bf a = load_frag(hb + lrow * 32 + koff);   // h chunk as A operand
        #pragma unroll
        for (int nt = 0; nt < 8; ++nt) {
          v16bf b = load_frag(w2s + (size_t)(nt * 16 + lrow) * 128 + kcl * 32 + koff);
          ff[nt] = __builtin_amdgcn_wmma_f32_16x16x32_bf16(false, a, false, b,
                                                           (short)0, ff[nt], false, false);
        }
        wait_ds();   // A/B-frag loads retired before hb is overwritten
      }
    }
    // All waves done with FFN weights; stage next layer's Wqkv under LN2.
    __syncthreads();
    if (l == 0)
      stage_issue(wstage, WqkvB + (size_t)384 * 128, 1, 98304, 98304, tid);

    #pragma unroll
    for (int nt = 0; nt < 8; ++nt)
      #pragma unroll
      for (int j = 0; j < 8; ++j)
        xf[(rbase + j) * 128 + nt * 16 + lrow] += ff[nt][j];   // residual
    wait_ds();
    layer_norm_rows(xf, xb, (l == 0 ? ln2_g : ln2_g + 128),
                    (l == 0 ? ln2_b : ln2_b + 128), lane);
    wait_ds();
    wait_async();
    __syncthreads();
  }

  // ---------------- emit CLS token (row 4*s) per sequence ----------------
  {
    const int c = lane * 4;
    #pragma unroll
    for (int s = 0; s < 4; ++s) {
      float4 v = *(const float4*)(xf + (s * 4) * 128 + c);
      *(float4*)(out + (size_t)(seq0 + s) * 128 + c) = v;
    }
  }
}

// ---------------- f32 -> bf16 weight conversion (into d_ws) ----------------
__global__ void convert_bf16_kernel(const float* __restrict__ src,
                                    uint16_t* __restrict__ dst, int n) {
  int i = blockIdx.x * blockDim.x + threadIdx.x;
  if (i < n) dst[i] = f2bf(src[i]);
}

extern "C" void kernel_launch(void* const* d_in, const int* in_sizes, int n_in,
                              void* d_out, int out_size, void* d_ws, size_t ws_size,
                              hipStream_t stream) {
  (void)in_sizes; (void)n_in; (void)out_size; (void)ws_size;
  const float* z0    = (const float*)d_in[0];
  const float* z1    = (const float*)d_in[1];
  const float* z2    = (const float*)d_in[2];
  const float* cls   = (const float*)d_in[3];
  const float* Wqkv  = (const float*)d_in[4];
  const float* bqkv  = (const float*)d_in[5];
  const float* Wo    = (const float*)d_in[6];
  const float* bo    = (const float*)d_in[7];
  const float* W1    = (const float*)d_in[8];
  const float* b1    = (const float*)d_in[9];
  const float* W2    = (const float*)d_in[10];
  const float* b2    = (const float*)d_in[11];
  const float* ln1_g = (const float*)d_in[12];
  const float* ln1_b = (const float*)d_in[13];
  const float* ln2_g = (const float*)d_in[14];
  const float* ln2_b = (const float*)d_in[15];
  float* out = (float*)d_out;

  const int nq = 2 * 384 * 128;   // Wqkv
  const int no = 2 * 128 * 128;   // Wo
  const int n1 = 2 * 512 * 128;   // W1
  const int n2 = 2 * 128 * 512;   // W2
  uint16_t* WqkvB = (uint16_t*)d_ws;
  uint16_t* WoB   = WqkvB + nq;
  uint16_t* W1B   = WoB + no;
  uint16_t* W2B   = W1B + n1;

  convert_bf16_kernel<<<(nq + 255) / 256, 256, 0, stream>>>(Wqkv, WqkvB, nq);
  convert_bf16_kernel<<<(no + 255) / 256, 256, 0, stream>>>(Wo,   WoB,   no);
  convert_bf16_kernel<<<(n1 + 255) / 256, 256, 0, stream>>>(W1,   W1B,   n1);
  convert_bf16_kernel<<<(n2 + 255) / 256, 256, 0, stream>>>(W2,   W2B,   n2);

  const int total_seq = 16 * 2048;               // B*T
  const int blocks    = total_seq / (WAVES * 4); // 1024
  epoch_mixer_kernel<<<dim3(blocks), dim3(256), SMEM_BYTES, stream>>>(
      z0, z1, z2, cls, WqkvB, bqkv, WoB, bo, W1B, b1, W2B, b2,
      ln1_g, ln1_b, ln2_g, ln2_b, out);
}
